// QBNNTransformer_21534966022690
// MI455X (gfx1250) — matrix operand order repaired and checked by reading, not compiled, not run on hardware
//
#include <hip/hip_runtime.h>
#include <hip/hip_bf16.h>

typedef __bf16 bf16;
typedef __attribute__((ext_vector_type(16))) __bf16 v16bf;
typedef __attribute__((ext_vector_type(8)))  __bf16 v8bf;
typedef __attribute__((ext_vector_type(8)))  float  v8f;

#define B_   4
#define S_   1024
#define D_   1024
#define H_   16
#define DK_  64
#define DFF_ 4096
#define NROW (B_ * S_)          // 4096
#define NEGV -1e9f

// ---------------------------------------------------------------------------
// WMMA fragment helpers (wave32, 16x16x32 bf16, A/B in this code's contiguous k order)
// A 16x32: lane l%16 = row M; lanes 0-15 hold K 0-15,
//          lanes 16-31 hold K 16-31 (contiguous along K).
// B 32x16 (as B^T row-major): lane n%16 = col N; lanes 0-15 hold K 0-15,
//          lanes 16-31 hold K 16-31 (contiguous along K in B^T rows).
// C/D 16x16 f32: VGPR r, lanes 0-15 -> (M=r, N=lane); lanes 16-31 -> (M=r+8).
// ---------------------------------------------------------------------------
static __device__ inline v16bf load_a_frag(const bf16* __restrict__ A, int lda,
                                           int row0, int k0, int lane) {
  int hf = lane >> 4;
  const bf16* p = A + (long long)(row0 + (lane & 15)) * lda + k0 + 16 * hf;
  v8bf lo = *reinterpret_cast<const v8bf*>(p);
  v8bf hi = *reinterpret_cast<const v8bf*>(p + 8);
  return __builtin_shufflevector(lo, hi, 0,1,2,3,4,5,6,7,8,9,10,11,12,13,14,15);
}

static __device__ inline v16bf load_b_frag(const bf16* __restrict__ Bt, int ldb,
                                           int n0, int k0, int lane) {
  int hf = lane >> 4;
  const bf16* p = Bt + (long long)(n0 + (lane & 15)) * ldb + k0 + 16 * hf;
  return *reinterpret_cast<const v16bf*>(p);
}

static __device__ inline v8f wmma_bf16(v16bf a, v16bf b, v8f c) {
  return __builtin_amdgcn_wmma_f32_16x16x32_bf16(false, a, false, b,
                                                 (short)0, c, false, false);
}

// ---------------------------------------------------------------------------
// Generic bf16 GEMM: C[m,n] = sum_k A[m,k] * Bt[n,k]   (Bt = B^T row-major)
// Each wave computes a 16x64 strip (4 wmma accumulators); 8 waves / block.
// MODE 0: bf16 out permuted [b,h,s,dk]   (+bias)      (Q,K projections)
// MODE 1: bf16 out permuted [b,h,dk,s]   (+bias)      (V transposed)
// MODE 2: f32 out  row-major             (+bias+res)  (Wo, FFN2)
// MODE 3: bf16 out row-major gelu*(1+0.1|sin 2θ|) (+bias)   (FFN1)
// MODE 4: bf16 out row-major ldc, batch offset        (attention P·V)
// ---------------------------------------------------------------------------
template <int MODE>
__global__ __launch_bounds__(256) void gemm_kernel(
    const bf16* __restrict__ A, const bf16* __restrict__ Bt,
    const float* __restrict__ bias, const float* __restrict__ res,
    const float* __restrict__ aux,
    float* __restrict__ outf, bf16* __restrict__ outb,
    int M, int N, int K, int lda, int ldb, int ldc,
    long long sAy, long long sBy, long long sOy,
    long long sAz, long long sBz, long long sOz)
{
  A  += (long long)blockIdx.y * sAy + (long long)blockIdx.z * sAz;
  Bt += (long long)blockIdx.y * sBy + (long long)blockIdx.z * sBz;
  long long obase = (long long)blockIdx.y * sOy + (long long)blockIdx.z * sOz;

  int tid  = threadIdx.x;
  int lane = tid & 31;
  int wave = tid >> 5;
  int ntN  = N >> 6;
  int wtile = blockIdx.x * 8 + wave;
  if (wtile >= (M >> 4) * ntN) return;
  int m0 = (wtile / ntN) << 4;
  int n0 = (wtile % ntN) << 6;

  v8f acc[4];
#pragma unroll
  for (int t = 0; t < 4; ++t) acc[t] = (v8f){0.f,0.f,0.f,0.f,0.f,0.f,0.f,0.f};

  for (int k0 = 0; k0 < K; k0 += 32) {
    if (k0 + 32 < K)  // gfx1250 speculative prefetch of next A chunk
      __builtin_prefetch(A + (long long)(m0 + (lane & 15)) * lda + k0 + 32, 0, 0);
    v16bf a = load_a_frag(A, lda, m0, k0, lane);
#pragma unroll
    for (int t = 0; t < 4; ++t) {
      v16bf b = load_b_frag(Bt, ldb, n0 + t * 16, k0, lane);
      acc[t] = wmma_bf16(a, b, acc[t]);
    }
  }

  int hf = lane >> 4, nl = lane & 15;
#pragma unroll
  for (int t = 0; t < 4; ++t) {
#pragma unroll
    for (int r = 0; r < 8; ++r) {
      int m = m0 + r + 8 * hf;
      int n = n0 + t * 16 + nl;
      float v = acc[t][r];
      if (MODE == 0) {
        v += bias[n];
        int bb = m >> 10, s = m & 1023, h = n >> 6, dk = n & 63;
        outb[((((long long)bb * H_ + h) * S_ + s) << 6) + dk] = (bf16)v;
      } else if (MODE == 1) {
        v += bias[n];
        int bb = m >> 10, s = m & 1023, h = n >> 6, dk = n & 63;
        outb[((((long long)bb * H_ + h) * DK_ + dk) << 10) + s] = (bf16)v;
      } else if (MODE == 2) {
        long long idx = (long long)m * ldc + n;
        outf[idx] = v + bias[n] + res[idx];
      } else if (MODE == 3) {
        v += bias[n];
        float g = 0.5f * v * (1.0f + erff(v * 0.70710678118654752f));
        float T = 1.0f + 0.1f * fabsf(sinf(2.0f * aux[n]));
        outb[(long long)m * ldc + n] = (bf16)(g * T);
      } else {
        outb[obase + (long long)m * ldc + n] = (bf16)v;
      }
    }
  }
}

// ---------------------------------------------------------------------------
// theta / r kernel: per row (b,h,s): th = sigmoid(q·Wt^T + bt)*pi/2,
// r = cos(2 th) (* strengths[h] for the q path).  One 64-thread block / row.
// ---------------------------------------------------------------------------
__global__ __launch_bounds__(64) void theta_kernel(
    const bf16* __restrict__ QK, const float* __restrict__ Wt,
    const float* __restrict__ bt, const float* __restrict__ strengths,
    int useS, bf16* __restrict__ R)
{
  long long row = blockIdx.x;
  int h = (int)((row >> 10) & (H_ - 1));
  int d = threadIdx.x;
  __shared__ float qs[DK_];
  const bf16* qr = QK + (row << 6);
  qs[d] = (float)qr[d];
  __syncthreads();
  float acc = bt[d];
  const float* wr = Wt + d * DK_;
#pragma unroll 8
  for (int e = 0; e < DK_; ++e) acc += qs[e] * wr[e];
  float th = (1.0f / (1.0f + expf(-acc))) * 1.57079632679489662f;
  float rv = cosf(2.0f * th);
  if (useS) rv *= strengths[h];
  R[(row << 6) + d] = (bf16)rv;
}

// ---------------------------------------------------------------------------
// Score kernel: block = 16 waves (one per head) for one (b, 16x16 i,j tile).
// Each wave: scores = q·k^T (wmma), qc = r_q'·r_k^T (wmma); qc tiles exchanged
// through LDS for the cross-head E mixing; causal mask; fully-masked tiles
// (jt > it) skipped.
// ---------------------------------------------------------------------------
__global__ __launch_bounds__(512) void scores_kernel(
    const bf16* __restrict__ Q, const bf16* __restrict__ Kb,
    const bf16* __restrict__ Rq, const bf16* __restrict__ Rk,
    const float* __restrict__ E, float* __restrict__ Sc)
{
  int jt = blockIdx.x, it = blockIdx.y, b = blockIdx.z;
  if (jt > it) return;
  __shared__ float qcb[H_][16][16];
  __shared__ float Es[H_ * H_];
  int tid = threadIdx.x;
  if (tid < H_ * H_) Es[tid] = E[tid];
  int h = tid >> 5, lane = tid & 31;
  long long headoff = (((long long)b * H_ + h) * S_) * DK_;
  const bf16* q  = Q  + headoff;
  const bf16* kk = Kb + headoff;
  const bf16* rq = Rq + headoff;
  const bf16* rk = Rk + headoff;
  int i0 = it << 4, j0 = jt << 4;
  v8f sacc = (v8f){0.f,0.f,0.f,0.f,0.f,0.f,0.f,0.f};
  v8f qacc = sacc;
#pragma unroll
  for (int k0 = 0; k0 < DK_; k0 += 32) {
    v16bf aq = load_a_frag(q,  DK_, i0, k0, lane);
    v16bf bk = load_b_frag(kk, DK_, j0, k0, lane);
    sacc = wmma_bf16(aq, bk, sacc);
    v16bf ar = load_a_frag(rq, DK_, i0, k0, lane);
    v16bf br = load_b_frag(rk, DK_, j0, k0, lane);
    qacc = wmma_bf16(ar, br, qacc);
  }
  int hf = lane >> 4, nl = lane & 15;
#pragma unroll
  for (int r = 0; r < 8; ++r) qcb[h][r + 8 * hf][nl] = qacc[r];
  __syncthreads();
#pragma unroll
  for (int r = 0; r < 8; ++r) {
    int i = r + 8 * hf, j = nl;
    float inter = 0.f;
#pragma unroll
    for (int g = 0; g < H_; ++g) inter += qcb[g][i][j] * Es[g * H_ + h];
    float v = sacc[r] * 0.125f + qacc[r] + 0.1f * inter;
    int gi = i0 + i, gj = j0 + j;
    if (gj > gi) v = NEGV;
    Sc[((((long long)b * H_ + h) * S_ + gi) << 10) + gj] = v;
  }
}

// ---------------------------------------------------------------------------
// Row softmax over 1024 with causal limit; fp32 scores in -> bf16 probs out.
// ---------------------------------------------------------------------------
__global__ __launch_bounds__(256) void softmax_kernel(
    const float* __restrict__ Sc, bf16* __restrict__ P)
{
  long long row = blockIdx.x;
  int i   = (int)(row & (S_ - 1));
  int tid = threadIdx.x;
  const float* srow = Sc + (row << 10);
  bf16* prow = P + (row << 10);
  __shared__ float red[256];
  float vals[4];
  float m = -3.4e38f;
#pragma unroll
  for (int p = 0; p < 4; ++p) {
    int j = tid + (p << 8);
    float v = (j <= i) ? srow[j] : -3.4e38f;
    vals[p] = v;
    m = fmaxf(m, v);
  }
  red[tid] = m; __syncthreads();
  for (int s = 128; s > 0; s >>= 1) {
    if (tid < s) red[tid] = fmaxf(red[tid], red[tid + s]);
    __syncthreads();
  }
  m = red[0]; __syncthreads();
  float sum = 0.f;
#pragma unroll
  for (int p = 0; p < 4; ++p) {
    int j = tid + (p << 8);
    float e = (j <= i) ? expf(vals[p] - m) : 0.f;
    vals[p] = e; sum += e;
  }
  red[tid] = sum; __syncthreads();
  for (int s = 128; s > 0; s >>= 1) {
    if (tid < s) red[tid] += red[tid + s];
    __syncthreads();
  }
  float inv = 1.0f / red[0];
#pragma unroll
  for (int p = 0; p < 4; ++p) prow[tid + (p << 8)] = (bf16)(vals[p] * inv);
}

// ---------------------------------------------------------------------------
// LayerNorm over D=1024; writes fp32 and (optionally) bf16 copies.
// ---------------------------------------------------------------------------
__global__ __launch_bounds__(256) void ln_kernel(
    const float* __restrict__ in, const float* __restrict__ g,
    const float* __restrict__ be, float* __restrict__ outf,
    bf16* __restrict__ outb)
{
  long long row = blockIdx.x;
  const float* x = in + row * D_;
  int tid = threadIdx.x;
  __shared__ float r1[256];
  __shared__ float r2[256];
  float v[4];
  float s = 0.f, s2 = 0.f;
#pragma unroll
  for (int p = 0; p < 4; ++p) {
    float t = x[tid + (p << 8)];
    v[p] = t; s += t; s2 += t * t;
  }
  r1[tid] = s; r2[tid] = s2; __syncthreads();
  for (int k = 128; k > 0; k >>= 1) {
    if (tid < k) { r1[tid] += r1[tid + k]; r2[tid] += r2[tid + k]; }
    __syncthreads();
  }
  float mu  = r1[0] * (1.0f / D_);
  float var = r2[0] * (1.0f / D_) - mu * mu;
  float rs  = rsqrtf(var + 1e-5f);
#pragma unroll
  for (int p = 0; p < 4; ++p) {
    int j = tid + (p << 8);
    float y = (v[p] - mu) * rs * g[j] + be[j];
    outf[row * D_ + j] = y;
    if (outb) outb[row * D_ + j] = (bf16)y;
  }
}

__global__ void cvt_kernel(const float* __restrict__ in, bf16* __restrict__ out,
                           long long n) {
  long long i = (long long)blockIdx.x * blockDim.x + threadIdx.x;
  long long stride = (long long)gridDim.x * blockDim.x;
  for (; i < n; i += stride) out[i] = (bf16)in[i];
}

// ---------------------------------------------------------------------------
extern "C" void kernel_launch(void* const* d_in, const int* in_sizes, int n_in,
                              void* d_out, int out_size, void* d_ws, size_t ws_size,
                              hipStream_t stream)
{
  const float* x        = (const float*)d_in[0];
  const float* Wq       = (const float*)d_in[2];
  const float* bq       = (const float*)d_in[3];
  const float* Wk       = (const float*)d_in[4];
  const float* bk       = (const float*)d_in[5];
  const float* Wv       = (const float*)d_in[6];
  const float* bv       = (const float*)d_in[7];
  const float* Wo       = (const float*)d_in[8];
  const float* bo       = (const float*)d_in[9];
  const float* Wt       = (const float*)d_in[10];
  const float* bt       = (const float*)d_in[11];
  const float* E        = (const float*)d_in[12];
  const float* strengths= (const float*)d_in[13];
  const float* W1       = (const float*)d_in[14];
  const float* b1       = (const float*)d_in[15];
  const float* W2       = (const float*)d_in[16];
  const float* b2       = (const float*)d_in[17];
  const float* theta_ff = (const float*)d_in[18];
  const float* g1       = (const float*)d_in[19];
  const float* beta1    = (const float*)d_in[20];
  const float* g3       = (const float*)d_in[21];
  const float* beta3    = (const float*)d_in[22];
  float* out            = (float*)d_out;

  char* ws = (char*)d_ws;
  size_t off = 0;
  auto alloc = [&](size_t bytes) -> char* {
    char* p = ws + off;
    off = (off + bytes + 255) & ~(size_t)255;
    return p;
  };

  const size_t ND2  = (size_t)NROW * D_ * 2;      // 8 MB (bf16 [4096,1024])
  const size_t DD2  = (size_t)D_ * D_ * 2;        // 2 MB
  const size_t FD2  = (size_t)DFF_ * D_ * 2;      // 8 MB
  const size_t SS4  = (size_t)B_*H_*S_*S_*4;      // 256 MB fp32 scores
  const size_t SS2  = (size_t)B_*H_*S_*S_*2;      // 128 MB bf16 probs
  const size_t ND4  = (size_t)NROW * D_ * 4;      // 16 MB fp32
  const size_t NF2  = (size_t)NROW * DFF_ * 2;    // 32 MB bf16 hidden

  bf16* xb   = (bf16*)alloc(ND2);
  bf16* Wqb  = (bf16*)alloc(DD2);
  bf16* Wkb  = (bf16*)alloc(DD2);
  bf16* Wvb  = (bf16*)alloc(DD2);
  bf16* Wob  = (bf16*)alloc(DD2);
  bf16* W1b  = (bf16*)alloc(FD2);
  bf16* W2b  = (bf16*)alloc(FD2);
  bf16* Qb   = (bf16*)alloc(ND2);   // [b,h,s,dk]
  bf16* Kbuf = (bf16*)alloc(ND2);   // [b,h,s,dk]
  bf16* Vtb  = (bf16*)alloc(ND2);   // [b,h,dk,s]
  bf16* Rqb  = (bf16*)alloc(ND2);
  bf16* Rkb  = (bf16*)alloc(ND2);
  bf16* attn = (bf16*)alloc(ND2);   // [b,s,h*dk]
  float* y   = (float*)alloc(ND4);  // post-Wo residual (reused for FFN2 out)
  float* x1f = (float*)alloc(ND4);
  bf16* x1b  = (bf16*)alloc(ND2);
  float* Sc  = (float*)alloc(SS4);  // fp32 pre-softmax scores (reused: hb)
  bf16* P    = (bf16*)alloc(SS2);   // bf16 probabilities
  bf16* hb   = (bf16*)Sc;           // FFN hidden aliases dead score buffer
  (void)hb; (void)NF2; (void)in_sizes; (void)n_in; (void)out_size; (void)ws_size;

  // 1) convert inputs/weights to bf16
  cvt_kernel<<<1024, 256, 0, stream>>>(x,  xb,  (long long)NROW * D_);
  cvt_kernel<<<512,  256, 0, stream>>>(Wq, Wqb, (long long)D_ * D_);
  cvt_kernel<<<512,  256, 0, stream>>>(Wk, Wkb, (long long)D_ * D_);
  cvt_kernel<<<512,  256, 0, stream>>>(Wv, Wvb, (long long)D_ * D_);
  cvt_kernel<<<512,  256, 0, stream>>>(Wo, Wob, (long long)D_ * D_);
  cvt_kernel<<<1024, 256, 0, stream>>>(W1, W1b, (long long)DFF_ * D_);
  cvt_kernel<<<1024, 256, 0, stream>>>(W2, W2b, (long long)DFF_ * D_);

  auto blocks = [](int M, int N) { return ((M / 16) * (N / 64) + 7) / 8; };

  // 2) QKV projections (head-major / transposed-V stores)
  gemm_kernel<0><<<dim3(blocks(NROW, D_)), 256, 0, stream>>>(
      xb, Wqb, bq, nullptr, nullptr, nullptr, Qb,
      NROW, D_, D_, D_, D_, 0, 0, 0, 0, 0, 0, 0);
  gemm_kernel<0><<<dim3(blocks(NROW, D_)), 256, 0, stream>>>(
      xb, Wkb, bk, nullptr, nullptr, nullptr, Kbuf,
      NROW, D_, D_, D_, D_, 0, 0, 0, 0, 0, 0, 0);
  gemm_kernel<1><<<dim3(blocks(NROW, D_)), 256, 0, stream>>>(
      xb, Wvb, bv, nullptr, nullptr, nullptr, Vtb,
      NROW, D_, D_, D_, D_, 0, 0, 0, 0, 0, 0, 0);

  // 3) quantum bias r vectors (strengths folded into r_q)
  theta_kernel<<<B_ * H_ * S_, 64, 0, stream>>>(Qb,   Wt, bt, strengths, 1, Rqb);
  theta_kernel<<<B_ * H_ * S_, 64, 0, stream>>>(Kbuf, Wt, bt, strengths, 0, Rkb);

  // 4) scores + quantum correction + E mixing + causal mask
  scores_kernel<<<dim3(S_ / 16, S_ / 16, B_), 512, 0, stream>>>(
      Qb, Kbuf, Rqb, Rkb, E, Sc);

  // 5) softmax -> bf16 probabilities
  softmax_kernel<<<B_ * H_ * S_, 256, 0, stream>>>(Sc, P);

  // 6) attention P·V (batched over b,h), out row-major [b,s,h*dk]
  gemm_kernel<4><<<dim3(8, H_, B_), 256, 0, stream>>>(
      P, Vtb, nullptr, nullptr, nullptr, nullptr, attn,
      S_, DK_, S_, S_, S_, D_,
      (long long)S_ * S_, (long long)DK_ * S_, (long long)DK_,
      (long long)H_ * S_ * S_, (long long)H_ * DK_ * S_, (long long)S_ * D_);

  // 7) output projection + residual, then LN1
  gemm_kernel<2><<<dim3(blocks(NROW, D_)), 256, 0, stream>>>(
      attn, Wob, bo, x, nullptr, y, nullptr,
      NROW, D_, D_, D_, D_, D_, 0, 0, 0, 0, 0, 0);
  ln_kernel<<<NROW, 256, 0, stream>>>(y, g1, beta1, x1f, x1b);

  // 8) FFN1 with exact GELU * (1 + 0.1|sin 2θ|)  (hidden aliases score buffer)
  gemm_kernel<3><<<dim3(blocks(NROW, DFF_)), 256, 0, stream>>>(
      x1b, W1b, b1, nullptr, theta_ff, nullptr, hb,
      NROW, DFF_, D_, D_, D_, DFF_, 0, 0, 0, 0, 0, 0);

  // 9) FFN2 + residual (reuse y), then LN3 -> d_out
  gemm_kernel<2><<<dim3(blocks(NROW, D_)), 256, 0, stream>>>(
      hb, W2b, b2, x1f, nullptr, y, nullptr,
      NROW, D_, DFF_, DFF_, DFF_, D_, 0, 0, 0, 0, 0, 0);
  ln_kernel<<<NROW, 256, 0, stream>>>(y, g3, beta3, out, nullptr);
}
